// PointerNet_47313359732785
// MI455X (gfx1250) — matrix operand order repaired and checked by reading, not compiled, not run on hardware
//
#include <hip/hip_runtime.h>
#include <hip/hip_bf16.h>
#include <math.h>

// ---------------- CDNA5 WMMA types ----------------
typedef __bf16 bf16;
typedef __attribute__((ext_vector_type(16))) __bf16 v16bf;
typedef __attribute__((ext_vector_type(8)))  __bf16 v8bf;
typedef __attribute__((ext_vector_type(8)))  float  v8f;

#define HDIM 128
#define NSEQ 256
#define BATCH 512

__device__ __forceinline__ float bf2f(bf16 x) { return (float)x; }
__device__ __forceinline__ bf16  f2bf(float x) { return (bf16)x; }
__device__ __forceinline__ float sigmoidf_(float x) { return 1.f / (1.f + __expf(-x)); }

// A fragment (16x32 bf16, M x K). contiguous k order, same as B (not the ISA table order):
// lane = g*16 + m ; element e: K = g*16 + e
__device__ __forceinline__ v16bf load_a_frag(const bf16* __restrict__ base,
                                             long rstride, int row0, int k0) {
  int lane = threadIdx.x & 31;
  int m = lane & 15, g = lane >> 4;
  const bf16* p = base + (long)(row0 + m) * rstride + k0 + g * 16;
  v8bf lo = *(const v8bf*)p;          // K = k0 + 16g + [0..7]
  v8bf hi = *(const v8bf*)(p + 8);    // K = k0 + 16g + 8 + [0..7]
  return __builtin_shufflevector(lo, hi, 0,1,2,3,4,5,6,7,8,9,10,11,12,13,14,15);
}

// B fragment (32x16 bf16, K x N) from PRE-TRANSPOSED weights wT[C][K=128]:
// lane = g*16 + n ; element e: K = g*16 + e  -> contiguous 32B per lane.
__device__ __forceinline__ v16bf load_b_frag(const bf16* __restrict__ wT,
                                             int col0, int k0) {
  int lane = threadIdx.x & 31;
  int n = lane & 15, g = lane >> 4;
  const bf16* p = wT + (long)(col0 + n) * HDIM + k0 + g * 16;
  return *(const v16bf*)p;
}

__device__ __forceinline__ v8f wmma_bf16(v16bf a, v16bf b, v8f c) {
  return __builtin_amdgcn_wmma_f32_16x16x32_bf16(false, a, false, b, (short)0, c,
                                                 false, false);
}

// ---------------- threefry2x32 (Gumbel sampling) ----------------
struct U2 { unsigned x, y; };
__device__ __forceinline__ unsigned rotl32(unsigned v, int s) {
  return (v << s) | (v >> (32 - s));
}
__device__ __forceinline__ U2 threefry2x32(U2 k, U2 c) {
  unsigned ks[3] = {k.x, k.y, 0x1BD11BDAu ^ k.x ^ k.y};
  unsigned x0 = c.x + ks[0], x1 = c.y + ks[1];
  const int R[8] = {13, 15, 26, 6, 17, 29, 16, 24};
#pragma unroll
  for (int i = 0; i < 5; i++) {
#pragma unroll
    for (int j = 0; j < 4; j++) { x0 += x1; x1 = rotl32(x1, R[(i * 4 + j) & 7]); x1 ^= x0; }
    x0 += ks[(i + 1) % 3];
    x1 += ks[(i + 2) % 3] + (unsigned)(i + 1);
  }
  return {x0, x1};
}

// ---------------- small prep kernels ----------------
__global__ void zero_kernel(float* p, int n) {
  int i = blockIdx.x * 256 + threadIdx.x;
  if (i < n) p[i] = 0.f;
}

__global__ void transpose_bf16_kernel(const float* __restrict__ src,
                                      bf16* __restrict__ dst, int K, int C) {
  int i = blockIdx.x * 256 + threadIdx.x;
  if (i >= K * C) return;
  int k = i / C, c = i - k * C;
  dst[(long)c * K + k] = f2bf(src[i]);
}

__global__ void fc_relu_kernel(const float* __restrict__ points,
                               const float* __restrict__ fc_w,
                               const float* __restrict__ fc_b,
                               bf16* __restrict__ xh) {
  long i = (long)blockIdx.x * 256 + threadIdx.x;     // B*N*H threads
  long row = i >> 7;
  int  h   = (int)(i & 127);
  float p0 = points[row * 2], p1 = points[row * 2 + 1];
  float v  = p0 * fc_w[h] + p1 * fc_w[HDIM + h] + fc_b[h];
  xh[i] = f2bf(fmaxf(v, 0.f));
}

__global__ void dec_input0_kernel(const float* __restrict__ points,
                                  const float* __restrict__ sp_w,
                                  const float* __restrict__ sp_b,
                                  bf16* __restrict__ dec_in) {
  int i = blockIdx.x * 256 + threadIdx.x;            // B*H threads
  int b = i >> 7, h = i & 127;
  const float* p = points + (long)b * NSEQ * 2;
  float s0 = 0.f, s1 = 0.f;
  for (int n = 0; n < NSEQ; n++) { s0 += p[2 * n]; s1 += p[2 * n + 1]; }
  s0 *= (1.f / NSEQ); s1 *= (1.f / NSEQ);
  dec_in[i] = f2bf(s0 * sp_w[h] + s1 * sp_w[HDIM + h] + sp_b[h]);
}

// ---------------- fused GRU step (both GEMMs + gates), WMMA ----------------
// 256 output tiles (32 m-tiles x 8 h-tiles) -> 32 blocks x 8 waves.
__global__ __launch_bounds__(256) void gru_step_kernel(
    const bf16* __restrict__ xb, long x_rs,
    const bf16* __restrict__ hbf, const float* __restrict__ hf,
    const bf16* __restrict__ wihT, const bf16* __restrict__ whhT,
    const float* __restrict__ bih, const float* __restrict__ bhh,
    float* __restrict__ h_out, bf16* __restrict__ hbf_out,
    bf16* __restrict__ extra_out, long extra_rs) {
  int wid  = threadIdx.x >> 5;
  int tile = blockIdx.x * 8 + wid;
  int row0 = (tile >> 3) * 16, col0 = (tile & 7) * 16;

  v8f z8 = {0.f, 0.f, 0.f, 0.f, 0.f, 0.f, 0.f, 0.f};
  v8f a_ir = z8, a_iz = z8, a_in = z8, a_hr = z8, a_hz = z8, a_hn = z8;
#pragma unroll
  for (int kc = 0; kc < HDIM; kc += 32) {
    v16bf ax = load_a_frag(xb, x_rs, row0, kc);
    v16bf ah = load_a_frag(hbf, HDIM, row0, kc);
    a_ir = wmma_bf16(ax, load_b_frag(wihT, col0,       kc), a_ir);
    a_iz = wmma_bf16(ax, load_b_frag(wihT, col0 + 128, kc), a_iz);
    a_in = wmma_bf16(ax, load_b_frag(wihT, col0 + 256, kc), a_in);
    a_hr = wmma_bf16(ah, load_b_frag(whhT, col0,       kc), a_hr);
    a_hz = wmma_bf16(ah, load_b_frag(whhT, col0 + 128, kc), a_hz);
    a_hn = wmma_bf16(ah, load_b_frag(whhT, col0 + 256, kc), a_hn);
  }
  int lane = threadIdx.x & 31;
  int n = lane & 15, g = lane >> 4;
  int col = col0 + n;
  float b_ir = bih[col], b_iz = bih[col + 128], b_in = bih[col + 256];
  float b_hr = bhh[col], b_hz = bhh[col + 128], b_hn = bhh[col + 256];
#pragma unroll
  for (int r = 0; r < 8; r++) {
    int row = row0 + r + 8 * g;        // C/D layout: VGPR r -> M = r + 8g
    float rr = sigmoidf_(a_ir[r] + b_ir + a_hr[r] + b_hr);
    float zz = sigmoidf_(a_iz[r] + b_iz + a_hz[r] + b_hz);
    float nn = tanhf(a_in[r] + b_in + rr * (a_hn[r] + b_hn));
    float hp = hf[(long)row * HDIM + col];
    float hn2 = (1.f - zz) * nn + zz * hp;
    h_out[(long)row * HDIM + col]   = hn2;
    hbf_out[(long)row * HDIM + col] = f2bf(hn2);
    if (extra_out) extra_out[(long)row * extra_rs + col] = f2bf(hn2);
  }
}

// ---------------- generic [rows,128]x[128,128] WMMA GEMMs ----------------
__global__ __launch_bounds__(256) void gemm128_bf16out_kernel(
    const bf16* __restrict__ A, long a_rs, const bf16* __restrict__ Bt,
    bf16* __restrict__ out) {
  int wid  = threadIdx.x >> 5;
  int tile = blockIdx.x * 8 + wid;
  int row0 = (tile >> 3) * 16, col0 = (tile & 7) * 16;
  v8f acc = {0.f, 0.f, 0.f, 0.f, 0.f, 0.f, 0.f, 0.f};
#pragma unroll
  for (int kc = 0; kc < HDIM; kc += 32)
    acc = wmma_bf16(load_a_frag(A, a_rs, row0, kc), load_b_frag(Bt, col0, kc), acc);
  int lane = threadIdx.x & 31;
  int n = lane & 15, g = lane >> 4;
#pragma unroll
  for (int r = 0; r < 8; r++)
    out[(long)(row0 + r + 8 * g) * HDIM + col0 + n] = f2bf(acc[r]);
}

__global__ __launch_bounds__(256) void gemm128_f32out_kernel(
    const bf16* __restrict__ A, long a_rs, const bf16* __restrict__ Bt,
    float* __restrict__ out) {
  int wid  = threadIdx.x >> 5;
  int tile = blockIdx.x * 8 + wid;
  int row0 = (tile >> 3) * 16, col0 = (tile & 7) * 16;
  v8f acc = {0.f, 0.f, 0.f, 0.f, 0.f, 0.f, 0.f, 0.f};
#pragma unroll
  for (int kc = 0; kc < HDIM; kc += 32)
    acc = wmma_bf16(load_a_frag(A, a_rs, row0, kc), load_b_frag(Bt, col0, kc), acc);
  int lane = threadIdx.x & 31;
  int n = lane & 15, g = lane >> 4;
#pragma unroll
  for (int r = 0; r < 8; r++)
    out[(long)(row0 + r + 8 * g) * HDIM + col0 + n] = acc[r];
}

// ---------------- attention scores + Gumbel-max sample + gather ----------------
__global__ __launch_bounds__(256) void score_sample_kernel(
    const bf16* __restrict__ w1e, const float* __restrict__ u,
    const float* __restrict__ vt, unsigned char* __restrict__ mask,
    const bf16* __restrict__ enc_outs, bf16* __restrict__ dec_next,
    float* __restrict__ seq_out, float* __restrict__ logp_out, int t) {
  __shared__ float s_u[HDIM], s_vt[HDIM], s_log[NSEQ], s_red[NSEQ];
  __shared__ int   s_idx[NSEQ];
  int b = blockIdx.x, n = threadIdx.x;
  if (n < HDIM) { s_u[n] = u[(long)b * HDIM + n]; s_vt[n] = vt[n]; }
  __syncthreads();

  float logit;
  if (mask[(long)b * NSEQ + n]) {
    logit = -1e30f;
  } else {
    const bf16* row = w1e + ((long)b * NSEQ + n) * HDIM;
    float s = 0.f;
#pragma unroll 4
    for (int h = 0; h < HDIM; h++) s += tanhf(bf2f(row[h]) + s_u[h]) * s_vt[h];
    logit = s;
  }
  s_log[n] = logit;

  // max
  s_red[n] = logit; __syncthreads();
  for (int off = 128; off > 0; off >>= 1) {
    if (n < off) s_red[n] = fmaxf(s_red[n], s_red[n + off]);
    __syncthreads();
  }
  float m = s_red[0]; __syncthreads();
  // sum exp
  s_red[n] = (logit < -1e29f) ? 0.f : __expf(logit - m); __syncthreads();
  for (int off = 128; off > 0; off >>= 1) {
    if (n < off) s_red[n] += s_red[n + off];
    __syncthreads();
  }
  float lse = m + __logf(s_red[0]); __syncthreads();

  // Gumbel-max categorical sample (threefry2x32 keyed on (t,b,n))
  U2 kt = threefry2x32(U2{0u, 42u}, U2{(unsigned)t, 0x9E3779B9u});
  U2 r2 = threefry2x32(kt, U2{(unsigned)b, (unsigned)n});
  float uu  = (r2.x >> 8) * (1.f / 16777216.f);
  float gum = -__logf(-__logf(fmaxf(uu, 1e-10f)));
  s_red[n] = logit + gum; s_idx[n] = n; __syncthreads();
  for (int off = 128; off > 0; off >>= 1) {
    if (n < off && s_red[n + off] > s_red[n]) {
      s_red[n] = s_red[n + off]; s_idx[n] = s_idx[n + off];
    }
    __syncthreads();
  }
  int idx = s_idx[0];
  if (n == 0) {
    seq_out[(long)b * NSEQ + t] = (float)idx;
    logp_out[b] += s_log[idx] - lse;
    mask[(long)b * NSEQ + idx] = 1;
  }
  __syncthreads();
  if (n < HDIM)
    dec_next[(long)b * HDIM + n] = enc_outs[((long)b * NSEQ + idx) * HDIM + n];
}

// ---------------- host launcher ----------------
extern "C" void kernel_launch(void* const* d_in, const int* in_sizes, int n_in,
                              void* d_out, int out_size, void* d_ws, size_t ws_size,
                              hipStream_t stream) {
  (void)in_sizes; (void)n_in; (void)out_size; (void)ws_size;
  const float* points  = (const float*)d_in[0];
  const float* fc_w    = (const float*)d_in[1];
  const float* fc_b    = (const float*)d_in[2];
  const float* enc_wih = (const float*)d_in[3];
  const float* enc_whh = (const float*)d_in[4];
  const float* enc_bih = (const float*)d_in[5];
  const float* enc_bhh = (const float*)d_in[6];
  const float* dec_wih = (const float*)d_in[7];
  const float* dec_whh = (const float*)d_in[8];
  const float* dec_bih = (const float*)d_in[9];
  const float* dec_bhh = (const float*)d_in[10];
  const float* w1      = (const float*)d_in[11];
  const float* w2      = (const float*)d_in[12];
  const float* vt      = (const float*)d_in[13];
  const float* sp_w    = (const float*)d_in[14];
  const float* sp_b    = (const float*)d_in[15];

  // ---- workspace carve (aligned 256B) ----
  size_t off = 0;
  char* base = (char*)d_ws;
  auto alloc = [&](size_t bytes) -> void* {
    off = (off + 255) & ~(size_t)255;
    void* p = base + off;
    off += bytes;
    return p;
  };
  const long BN  = (long)BATCH * NSEQ;          // 131072
  const long BNH = BN * HDIM;                   // 16,777,216
  bf16* xh       = (bf16*)alloc(BNH * 2);       // relu(fc) activations
  bf16* enc_outs = (bf16*)alloc(BNH * 2);       // encoder hidden states
  bf16* w1e      = (bf16*)alloc(BNH * 2);       // enc_outs @ w1 (L2-resident, 33.5MB)
  bf16* wihT_e   = (bf16*)alloc(384 * 128 * 2);
  bf16* whhT_e   = (bf16*)alloc(384 * 128 * 2);
  bf16* wihT_d   = (bf16*)alloc(384 * 128 * 2);
  bf16* whhT_d   = (bf16*)alloc(384 * 128 * 2);
  bf16* w1T      = (bf16*)alloc(128 * 128 * 2);
  bf16* w2T      = (bf16*)alloc(128 * 128 * 2);
  float* h_a     = (float*)alloc((long)BATCH * HDIM * 4);
  float* h_b     = (float*)alloc((long)BATCH * HDIM * 4);
  bf16* hbf_a    = (bf16*)alloc((long)BATCH * HDIM * 2);
  bf16* hbf_b    = (bf16*)alloc((long)BATCH * HDIM * 2);
  bf16* din_a    = (bf16*)alloc((long)BATCH * HDIM * 2);
  bf16* din_b    = (bf16*)alloc((long)BATCH * HDIM * 2);
  float* u       = (float*)alloc((long)BATCH * HDIM * 4);
  unsigned char* mask = (unsigned char*)alloc(BN);

  float* seq_out  = (float*)d_out;              // [B,N] indices as float
  float* logp_out = (float*)d_out + BN;         // [B]

  // ---- init (deterministic per call) ----
  zero_kernel<<<(32768 + 255) / 256, 256, 0, stream>>>((float*)mask, 32768);
  zero_kernel<<<(512 + 255) / 256, 256, 0, stream>>>(logp_out, 512);
  zero_kernel<<<(65536 + 255) / 256, 256, 0, stream>>>(h_a, 65536);
  zero_kernel<<<(32768 + 255) / 256, 256, 0, stream>>>((float*)hbf_a, 32768);

  // ---- weight convert + transpose to bf16 [C][K] ----
  transpose_bf16_kernel<<<(49152 + 255) / 256, 256, 0, stream>>>(enc_wih, wihT_e, 128, 384);
  transpose_bf16_kernel<<<(49152 + 255) / 256, 256, 0, stream>>>(enc_whh, whhT_e, 128, 384);
  transpose_bf16_kernel<<<(49152 + 255) / 256, 256, 0, stream>>>(dec_wih, wihT_d, 128, 384);
  transpose_bf16_kernel<<<(49152 + 255) / 256, 256, 0, stream>>>(dec_whh, whhT_d, 128, 384);
  transpose_bf16_kernel<<<(16384 + 255) / 256, 256, 0, stream>>>(w1, w1T, 128, 128);
  transpose_bf16_kernel<<<(16384 + 255) / 256, 256, 0, stream>>>(w2, w2T, 128, 128);

  // ---- fc + relu ----
  fc_relu_kernel<<<(unsigned)(BNH / 256), 256, 0, stream>>>(points, fc_w, fc_b, xh);

  // ---- encoder GRU scan (fused WMMA step kernel per t) ----
  float* hf_cur = h_a;  float* hf_nxt = h_b;
  bf16* hb_cur  = hbf_a; bf16* hb_nxt = hbf_b;
  for (int t = 0; t < NSEQ; t++) {
    gru_step_kernel<<<32, 256, 0, stream>>>(
        xh + (long)t * HDIM, (long)NSEQ * HDIM, hb_cur, hf_cur,
        wihT_e, whhT_e, enc_bih, enc_bhh,
        hf_nxt, hb_nxt, enc_outs + (long)t * HDIM, (long)NSEQ * HDIM);
    float* tf = hf_cur; hf_cur = hf_nxt; hf_nxt = tf;
    bf16*  tb = hb_cur; hb_cur = hb_nxt; hb_nxt = tb;
  }

  // ---- w1e = enc_outs @ w1 (big WMMA GEMM, 131072x128x128) ----
  gemm128_bf16out_kernel<<<8192, 256, 0, stream>>>(enc_outs, HDIM, w1T, w1e);

  // ---- decoder initial input ----
  dec_input0_kernel<<<(65536 + 255) / 256, 256, 0, stream>>>(points, sp_w, sp_b, din_a);

  // ---- decoder scan ----
  bf16* di_cur = din_a; bf16* di_nxt = din_b;
  for (int t = 0; t < NSEQ; t++) {
    gru_step_kernel<<<32, 256, 0, stream>>>(
        di_cur, HDIM, hb_cur, hf_cur,
        wihT_d, whhT_d, dec_bih, dec_bhh,
        hf_nxt, hb_nxt, nullptr, 0);
    gemm128_f32out_kernel<<<32, 256, 0, stream>>>(hb_nxt, HDIM, w2T, u);
    score_sample_kernel<<<BATCH, 256, 0, stream>>>(
        w1e, u, vt, mask, enc_outs, di_nxt, seq_out, logp_out, t);
    float* tf = hf_cur; hf_cur = hf_nxt; hf_nxt = tf;
    bf16*  tb = hb_cur; hb_cur = hb_nxt; hb_nxt = tb;
    bf16*  td = di_cur; di_cur = di_nxt; di_nxt = td;
  }
}
